// T5Attention_6193342841648
// MI455X (gfx1250) — matrix operand order repaired and checked by reading, not compiled, not run on hardware
//
#include <hip/hip_runtime.h>
#include <hip/hip_bf16.h>

// ---------------------------------------------------------------------------
// Types for CDNA5 WMMA (wave32): v_wmma_f32_16x16x32_bf16
// ---------------------------------------------------------------------------
typedef __bf16 bf16_t;
typedef __attribute__((ext_vector_type(8)))  bf16_t v8bf;
typedef __attribute__((ext_vector_type(16))) bf16_t v16bf;
typedef __attribute__((ext_vector_type(8)))  float  v8f;
typedef __attribute__((ext_vector_type(4)))  unsigned v4u;
typedef __attribute__((ext_vector_type(4)))  int      v4i;
typedef __attribute__((ext_vector_type(8)))  int      v8i;

#define WMMA_BF16(a, b, c) \
  __builtin_amdgcn_wmma_f32_16x16x32_bf16(false, (a), false, (b), (short)0, (c), false, false)

__device__ __forceinline__ v16bf concat8(v8bf lo, v8bf hi) {
  return __builtin_shufflevector(lo, hi, 0,1,2,3,4,5,6,7,8,9,10,11,12,13,14,15);
}

// A fragment (16x32 bf16, M x K): lane r = lane&15 is the row; half hl = lane>>4.
// VGPR0-3 hold K = hl*8 + 0..7, VGPR4-7 hold K = 16 + hl*8 + 0..7.
// `base` points at element (row r, k = 0); row pitch is 32 elements for LDS tiles.
__device__ __forceinline__ v16bf load_frag_a(const bf16_t* base, int hl) {
  v8bf lo = *(const v8bf*)(base + hl * 8);
  v8bf hi = *(const v8bf*)(base + 16 + hl * 8);
  return concat8(lo, hi);
}

// B fragment (32x16 bf16, K x N): lane holds column c = lane&15; half hl = lane>>4
// holds K = hl*8 + 0..7 and 16 + hl*8 + 0..7. Source is N-major; `base` points at (column c, k = 0).
__device__ __forceinline__ v16bf load_frag_b(const bf16_t* base, int hl) {
  v8bf lo = *(const v8bf*)(base + hl * 8);
  v8bf hi = *(const v8bf*)(base + 16 + hl * 8);
  return concat8(lo, hi);
}

// ---------------------------------------------------------------------------
// Tensor Data Mover: 2D tile (tile_rows x tile_k bf16 elems) global -> LDS.
// Descriptor bitfields per CDNA5 ISA.
// This toolchain exposes the 6-arg builtin:
//   (v4u g0, v8i g1, v4i g2, v4i g3, v8i g4, i32 cpol)
// ---------------------------------------------------------------------------
__device__ __forceinline__ void tdm_load_tile_2d(unsigned lds_addr, const void* gptr,
                                                 unsigned tile_k, unsigned tile_rows,
                                                 unsigned stride_elems, unsigned tensor_rows) {
  unsigned long long ga = (unsigned long long)(uintptr_t)gptr;
  v4u g0 = { 1u,                                   // count=1, user-mode, no gather
             lds_addr,                             // LDS byte address
             (unsigned)ga,                         // global_addr[31:0]
             (unsigned)((ga >> 32) & 0x01ffffffu)  // global_addr[56:32]
               | (2u << 30) };                     // type=2 ("image")
  const unsigned td0 = stride_elems;               // tensor_dim0 (row length)
  const unsigned td1 = tensor_rows;                // tensor_dim1 (rows)
  v8i g1;
  g1[0] = (int)(1u << 16);                         // wg_mask=0, data_size=1 (2B)
  g1[1] = (int)((td0 & 0xffffu) << 16);            // atomic_addr=0 | tensor_dim0[15:0]
  g1[2] = (int)(((td0 >> 16) & 0xffffu) | ((td1 & 0xffffu) << 16));
  g1[3] = (int)(((td1 >> 16) & 0xffffu) | ((tile_k & 0xffffu) << 16));   // tile_dim0
  g1[4] = (int)(tile_rows & 0xffffu);              // tile_dim1 | tile_dim2=0
  g1[5] = (int)stride_elems;                       // tensor_dim0_stride[31:0]
  g1[6] = 0;                                       // stride[47:32] | dim1_stride lo
  g1[7] = 0;
  v4i gz4 = {0, 0, 0, 0};
  v8i gz8 = {0, 0, 0, 0, 0, 0, 0, 0};
  __builtin_amdgcn_tensor_load_to_lds(g0, g1, gz4, gz4, gz8, 0);
}

// ---------------------------------------------------------------------------
// Problem constants
// ---------------------------------------------------------------------------
#define BATCH 2
#define SEQ   2048
#define DMOD  1024
#define HEADS 16
#define DH    64
#define ROWS  (BATCH * SEQ)          // 4096

// Workspace byte offsets (Ctx reuses xb: x is dead after the QKV GEMM)
#define OFF_XB   (size_t)0
#define OFF_WQT  (size_t)(8u  * 1024 * 1024)
#define OFF_WKT  (size_t)(10u * 1024 * 1024)
#define OFF_WVT  (size_t)(12u * 1024 * 1024)
#define OFF_WOT  (size_t)(14u * 1024 * 1024)
#define OFF_Q    (size_t)(16u * 1024 * 1024)
#define OFF_K    (size_t)(24u * 1024 * 1024)
#define OFF_VT   (size_t)(32u * 1024 * 1024)
#define OFF_CTX  OFF_XB

// ---------------------------------------------------------------------------
// Kernel 1: fp32 -> bf16 (grid-stride)
// ---------------------------------------------------------------------------
__global__ void cvt_f32_to_bf16(const float* __restrict__ src, bf16_t* __restrict__ dst, int n) {
  int i = blockIdx.x * blockDim.x + threadIdx.x;
  int stride = gridDim.x * blockDim.x;
  for (; i < n; i += stride) dst[i] = (bf16_t)src[i];
}

// ---------------------------------------------------------------------------
// Kernel 2: W[k][n] fp32 -> Wt[n][k] bf16 (LDS tiled transpose, 1024x1024)
// ---------------------------------------------------------------------------
__global__ void transpose_to_bf16(const float* __restrict__ W, bf16_t* __restrict__ Wt) {
  __shared__ float tile[32][33];
  const int x0 = blockIdx.x * 32, y0 = blockIdx.y * 32;
  const int tx = threadIdx.x;
  for (int j = threadIdx.y; j < 32; j += 8)
    tile[j][tx] = W[(size_t)(y0 + j) * DMOD + x0 + tx];
  __syncthreads();
  for (int j = threadIdx.y; j < 32; j += 8)
    Wt[(size_t)(x0 + j) * DMOD + y0 + tx] = (bf16_t)tile[tx][j];
}

// ---------------------------------------------------------------------------
// Shared GEMM body: WG tile 128x128, K-step 32, TDM double-buffered LDS,
// 8 waves each computing a 32x64 register tile (2x4 accumulators).
// Layout inside LDS tiles: [localRow][32] bf16, row pitch 64 B.
//   lA rows  = M rows (A row-major),  lB rows = N columns (B stored N-major).
// ---------------------------------------------------------------------------
#define GEMM_BODY(Abase, Bbase, acc, lA, lB, wave, waveM, waveN, r, hl)                      \
  do {                                                                                       \
    const int NIT = DMOD / 32;                                                               \
    if ((wave) == 0) {                                                                       \
      tdm_load_tile_2d((unsigned)(uintptr_t)&lA[0][0], (Abase), 32, 128, DMOD, 128);         \
      tdm_load_tile_2d((unsigned)(uintptr_t)&lB[0][0], (Bbase), 32, 128, DMOD, 128);         \
    }                                                                                        \
    for (int it = 0; it < NIT; ++it) {                                                       \
      const int buf = it & 1;                                                                \
      if ((wave) == 0) {                                                                     \
        if (it + 1 < NIT) {                                                                  \
          tdm_load_tile_2d((unsigned)(uintptr_t)&lA[buf ^ 1][0], (Abase) + (it + 1) * 32,    \
                           32, 128, DMOD, 128);                                              \
          tdm_load_tile_2d((unsigned)(uintptr_t)&lB[buf ^ 1][0], (Bbase) + (it + 1) * 32,    \
                           32, 128, DMOD, 128);                                              \
          __builtin_amdgcn_s_wait_tensorcnt(2);  /* in-order: current buffer done */         \
        } else {                                                                             \
          __builtin_amdgcn_s_wait_tensorcnt(0);                                              \
        }                                                                                    \
      }                                                                                      \
      __syncthreads(); /* buffer `buf` visible to all waves */                               \
      const bf16_t* As = &lA[buf][0] + (size_t)((waveM) * 32) * 32;                          \
      const bf16_t* Bs = &lB[buf][0] + (size_t)((waveN) * 64) * 32;                          \
      v16bf a0 = load_frag_a(As + (size_t)(r)*32, (hl));                                     \
      v16bf a1 = load_frag_a(As + (size_t)(16 + (r)) * 32, (hl));                            \
      _Pragma("unroll")                                                                      \
      for (int nf = 0; nf < 4; ++nf) {                                                       \
        v16bf b = load_frag_b(Bs + (size_t)(nf * 16 + (r)) * 32, (hl));                      \
        acc[0][nf] = WMMA_BF16(a0, b, acc[0][nf]);                                           \
        acc[1][nf] = WMMA_BF16(a1, b, acc[1][nf]);                                           \
      }                                                                                      \
      __syncthreads(); /* reads done before this buffer is overwritten */                    \
    }                                                                                        \
  } while (0)

// ---------------------------------------------------------------------------
// Kernel 3: fused QKV projection (blockIdx.y selects Q/K/V).
// ---------------------------------------------------------------------------
__global__ void __launch_bounds__(256)
gemm_qkv_kernel(const bf16_t* __restrict__ xb,
                const bf16_t* __restrict__ Wqt, const bf16_t* __restrict__ Wkt,
                const bf16_t* __restrict__ Wvt,
                bf16_t* __restrict__ Qb, bf16_t* __restrict__ Kb, bf16_t* __restrict__ Vt) {
  __shared__ __align__(16) bf16_t lA[2][128 * 32];
  __shared__ __align__(16) bf16_t lB[2][128 * 32];

  const int which = blockIdx.y;
  const bf16_t* Wt = (which == 0) ? Wqt : (which == 1) ? Wkt : Wvt;

  const int row0 = (blockIdx.x >> 3) * 128;   // 32 M-tiles
  const int col0 = (blockIdx.x & 7) * 128;    // 8  N-tiles
  const int wave = threadIdx.x >> 5, lane = threadIdx.x & 31;
  const int r = lane & 15, hl = lane >> 4;
  const int waveM = wave >> 1, waveN = wave & 1;   // 4 x 2 wave grid

  v8f zero = {0.f,0.f,0.f,0.f,0.f,0.f,0.f,0.f};
  v8f acc[2][4] = {{zero, zero, zero, zero}, {zero, zero, zero, zero}};

  const bf16_t* Abase = xb + (size_t)row0 * DMOD;
  const bf16_t* Bbase = Wt + (size_t)col0 * DMOD;
  GEMM_BODY(Abase, Bbase, acc, lA, lB, wave, waveM, waveN, r, hl);

#pragma unroll
  for (int sub = 0; sub < 2; ++sub)
#pragma unroll
    for (int nf = 0; nf < 4; ++nf)
#pragma unroll
      for (int i = 0; i < 8; ++i) {
        const int row = row0 + waveM * 32 + sub * 16 + i + 8 * hl;
        const int col = col0 + waveN * 64 + nf * 16 + r;
        const float v = acc[sub][nf][i];
        if (which == 0) {
          Qb[(size_t)row * DMOD + col] = (bf16_t)(v * 0.125f);   // dh^-0.5
        } else if (which == 1) {
          Kb[(size_t)row * DMOD + col] = (bf16_t)v;
        } else {
          const int b = row >> 11, nn = row & (SEQ - 1);
          const int h = col >> 6, dh = col & (DH - 1);
          Vt[(((size_t)(b * HEADS + h) * DH + dh) << 11) + nn] = (bf16_t)v;
        }
      }
}

// ---------------------------------------------------------------------------
// Kernel 4: flash attention with T5 relative-position bias + causal mask.
// ---------------------------------------------------------------------------
__global__ void __launch_bounds__(128)
attn_kernel(const bf16_t* __restrict__ Qb, const bf16_t* __restrict__ Kb,
            const bf16_t* __restrict__ Vt, const float* __restrict__ rel_emb,
            bf16_t* __restrict__ Ctx) {
  const int bh = blockIdx.y, batch = bh >> 4, head = bh & 15;
  const int q0 = blockIdx.x * 64;

  __shared__ float biasS[32];
  __shared__ __align__(16) bf16_t Pld[4][16 * 32];
  if (threadIdx.x < 32) biasS[threadIdx.x] = rel_emb[threadIdx.x * HEADS + head];
  __syncthreads();

  const int wave = threadIdx.x >> 5, lane = threadIdx.x & 31;
  const int r = lane & 15, hl = lane >> 4;
  const int qw = q0 + wave * 16;

  const bf16_t* qrow = Qb + (size_t)(batch * SEQ + qw + r) * DMOD + head * DH;
  const v16bf aq0 = load_frag_a(qrow, hl);
  const v16bf aq1 = load_frag_a(qrow + 32, hl);

  v8f zero = {0.f,0.f,0.f,0.f,0.f,0.f,0.f,0.f};
  v8f o[4] = {zero, zero, zero, zero};
  float m_i[8], l_i[8];
#pragma unroll
  for (int i = 0; i < 8; ++i) { m_i[i] = -1e30f; l_i[i] = 0.f; }

  const float INV_LOG8_X16 = 7.6943636f;
  bf16_t* pt = &Pld[wave][0];
  const bf16_t* vbase0 = Vt + ((size_t)(batch * HEADS + head) * DH) * (size_t)SEQ;

  for (int kb = 0; kb < q0 + 64; kb += 32) {
    v8f s[2] = {zero, zero};
#pragma unroll
    for (int sh = 0; sh < 2; ++sh) {
      const bf16_t* krow = Kb + (size_t)(batch * SEQ + kb + sh * 16 + r) * DMOD + head * DH;
      s[sh] = WMMA_BF16(aq0, load_frag_b(krow, hl),      s[sh]);
      s[sh] = WMMA_BF16(aq1, load_frag_b(krow + 32, hl), s[sh]);
    }

    float alpha[8];
#pragma unroll
    for (int i = 0; i < 8; ++i) {
      const int q = qw + i + 8 * hl;
#pragma unroll
      for (int sh = 0; sh < 2; ++sh) {
        const int key = kb + sh * 16 + r;
        float v = s[sh][i];
        if (key > q) {
          v = -1e30f;
        } else {
          const int nneg = q - key;
          int bucket;
          if (nneg < 16) bucket = nneg;
          else {
            bucket = 16 + (int)(__logf((float)nneg * 0.0625f) * INV_LOG8_X16);
            if (bucket > 31) bucket = 31;
          }
          v += biasS[bucket];
        }
        s[sh][i] = v;
      }
      float rm = fmaxf(s[0][i], s[1][i]);
      rm = fmaxf(rm, __shfl_xor(rm, 1, 32));
      rm = fmaxf(rm, __shfl_xor(rm, 2, 32));
      rm = fmaxf(rm, __shfl_xor(rm, 4, 32));
      rm = fmaxf(rm, __shfl_xor(rm, 8, 32));
      const float mn = fmaxf(m_i[i], rm);
      alpha[i] = __expf(m_i[i] - mn);
      m_i[i] = mn;
      const float p0 = __expf(s[0][i] - mn);
      const float p1 = __expf(s[1][i] - mn);
      s[0][i] = p0; s[1][i] = p1;
      float rs = p0 + p1;
      rs += __shfl_xor(rs, 1, 32);
      rs += __shfl_xor(rs, 2, 32);
      rs += __shfl_xor(rs, 4, 32);
      rs += __shfl_xor(rs, 8, 32);
      l_i[i] = l_i[i] * alpha[i] + rs;
    }
#pragma unroll
    for (int nf = 0; nf < 4; ++nf)
#pragma unroll
      for (int i = 0; i < 8; ++i) o[nf][i] *= alpha[i];

    // C-fragment -> A-fragment re-layout through wave-private LDS tile
#pragma unroll
    for (int sh = 0; sh < 2; ++sh)
#pragma unroll
      for (int i = 0; i < 8; ++i)
        pt[(i + 8 * hl) * 32 + sh * 16 + r] = (bf16_t)s[sh][i];
    __syncthreads();
    const v8bf plo = *(const v8bf*)(pt + r * 32 + hl * 8);
    const v8bf phi = *(const v8bf*)(pt + r * 32 + 16 + hl * 8);
    const v16bf ap = concat8(plo, phi);
    __syncthreads();

    const bf16_t* vb = vbase0 + kb + hl * 8;
#pragma unroll
    for (int nf = 0; nf < 4; ++nf) {
      const bf16_t* vrow = vb + (size_t)(nf * 16 + r) * SEQ;
      v16bf bv = concat8(*(const v8bf*)vrow, *(const v8bf*)(vrow + 16));
      o[nf] = WMMA_BF16(ap, bv, o[nf]);
    }
  }

#pragma unroll
  for (int i = 0; i < 8; ++i) {
    const float inv = 1.0f / l_i[i];
    const size_t row = (size_t)(batch * SEQ + qw + i + 8 * hl);
#pragma unroll
    for (int nf = 0; nf < 4; ++nf)
      Ctx[row * DMOD + head * DH + nf * 16 + r] = (bf16_t)(o[nf][i] * inv);
  }
}

// ---------------------------------------------------------------------------
// Kernel 5: out = Ctx @ Wo + bo (fp32 output), same TDM-pipelined body.
// ---------------------------------------------------------------------------
__global__ void __launch_bounds__(256)
gemm_out_kernel(const bf16_t* __restrict__ Ctx, const bf16_t* __restrict__ Wot,
                const float* __restrict__ bo, float* __restrict__ out) {
  __shared__ __align__(16) bf16_t lA[2][128 * 32];
  __shared__ __align__(16) bf16_t lB[2][128 * 32];

  const int row0 = (blockIdx.x >> 3) * 128;
  const int col0 = (blockIdx.x & 7) * 128;
  const int wave = threadIdx.x >> 5, lane = threadIdx.x & 31;
  const int r = lane & 15, hl = lane >> 4;
  const int waveM = wave >> 1, waveN = wave & 1;

  v8f zero = {0.f,0.f,0.f,0.f,0.f,0.f,0.f,0.f};
  v8f acc[2][4] = {{zero, zero, zero, zero}, {zero, zero, zero, zero}};

  const bf16_t* Abase = Ctx + (size_t)row0 * DMOD;
  const bf16_t* Bbase = Wot + (size_t)col0 * DMOD;
  GEMM_BODY(Abase, Bbase, acc, lA, lB, wave, waveM, waveN, r, hl);

#pragma unroll
  for (int sub = 0; sub < 2; ++sub)
#pragma unroll
    for (int nf = 0; nf < 4; ++nf)
#pragma unroll
      for (int i = 0; i < 8; ++i) {
        const int row = row0 + waveM * 32 + sub * 16 + i + 8 * hl;
        const int col = col0 + waveN * 64 + nf * 16 + r;
        out[(size_t)row * DMOD + col] = acc[sub][nf][i] + bo[col];
      }
}

// ---------------------------------------------------------------------------
// Host launcher
// ---------------------------------------------------------------------------
extern "C" void kernel_launch(void* const* d_in, const int* in_sizes, int n_in,
                              void* d_out, int out_size, void* d_ws, size_t ws_size,
                              hipStream_t stream) {
  const float* x       = (const float*)d_in[0];
  const float* Wq      = (const float*)d_in[1];
  const float* Wk      = (const float*)d_in[2];
  const float* Wv      = (const float*)d_in[3];
  const float* Wo      = (const float*)d_in[4];
  const float* bo      = (const float*)d_in[5];
  const float* rel_emb = (const float*)d_in[6];

  char* ws = (char*)d_ws;
  bf16_t* xb  = (bf16_t*)(ws + OFF_XB);
  bf16_t* Wqt = (bf16_t*)(ws + OFF_WQT);
  bf16_t* Wkt = (bf16_t*)(ws + OFF_WKT);
  bf16_t* Wvt = (bf16_t*)(ws + OFF_WVT);
  bf16_t* Wot = (bf16_t*)(ws + OFF_WOT);
  bf16_t* Qb  = (bf16_t*)(ws + OFF_Q);
  bf16_t* Kb  = (bf16_t*)(ws + OFF_K);
  bf16_t* Vt  = (bf16_t*)(ws + OFF_VT);
  bf16_t* Ctx = (bf16_t*)(ws + OFF_CTX);

  // 1) x -> bf16
  cvt_f32_to_bf16<<<1024, 256, 0, stream>>>(x, xb, ROWS * DMOD);

  // 2) weights -> transposed bf16
  {
    dim3 tb(32, 8), tg(DMOD / 32, DMOD / 32);
    transpose_to_bf16<<<tg, tb, 0, stream>>>(Wq, Wqt);
    transpose_to_bf16<<<tg, tb, 0, stream>>>(Wk, Wkt);
    transpose_to_bf16<<<tg, tb, 0, stream>>>(Wv, Wvt);
    transpose_to_bf16<<<tg, tb, 0, stream>>>(Wo, Wot);
  }

  // 3) fused QKV projections: 128x128 WG tiles, TDM double-buffered
  gemm_qkv_kernel<<<dim3((ROWS / 128) * (DMOD / 128), 3), 256, 0, stream>>>(
      xb, Wqt, Wkt, Wvt, Qb, Kb, Vt);

  // 4) flash attention with T5 bias
  attn_kernel<<<dim3(SEQ / 64, BATCH * HEADS), 128, 0, stream>>>(Qb, Kb, Vt, rel_emb, Ctx);

  // 5) output projection + bias
  gemm_out_kernel<<<(ROWS / 128) * (DMOD / 128), 256, 0, stream>>>(
      Ctx, Wot, bo, (float*)d_out);
}
